// HTMModel_30090540876452
// MI455X (gfx1250) — matrix operand order repaired and checked by reading, not compiled, not run on hardware
//
#include <hip/hip_runtime.h>
#include <hip/hip_bf16.h>

typedef __attribute__((ext_vector_type(16))) _Float16 v16h;
typedef __attribute__((ext_vector_type(8)))  _Float16 v8h;
typedef __attribute__((ext_vector_type(4)))  _Float16 v4h;
typedef __attribute__((ext_vector_type(8)))  float    v8f;
typedef __attribute__((ext_vector_type(4)))  float    v4f;

#define D_IN   4096   // reduction dim
#define N_COLS 2048   // columns (k in einsum)
#define M_ROWS 4096   // 2 * 2048 flattened (b,s)
#define TILE_M 256
#define TILE_N 128
#define TILE_K 32
#define KSTEPS (D_IN / TILE_K)
#define LDSS   40     // LDS row stride in halves (padded 32 -> 40, conflict-free b128)

// ---------------------------------------------------------------------------
// Kernel 1: C[m,n] = boost[n] * sum_d A[m,d] * B[n,d]   (A: 4096x4096, B: 2048x4096)
// f32 -> f16 convert on the fly, f32 accumulate via v_wmma_f32_16x16x32_f16.
// Block tile 256x128, 8 wave32 in a 4x2 grid, 64x64 per wave (16 WMMA tiles).
// Double-buffered LDS: one barrier per K-step; global prefetch 2 slabs ahead.
// ---------------------------------------------------------------------------
__global__ __launch_bounds__(256)
void htm_overlap_gemm(const float* __restrict__ A,
                      const float* __restrict__ Bm,
                      const float* __restrict__ boost,
                      float* __restrict__ C)
{
    __shared__ _Float16 As[2][TILE_M * LDSS];   // 2 x 20 KB
    __shared__ _Float16 Bs[2][TILE_N * LDSS];   // 2 x 10 KB

    const int tid  = threadIdx.x;
    const int lane = tid & 31;
    const int wave = tid >> 5;          // 0..7
    const int wm   = wave >> 1;         // 0..3 : 64-row slice of tile
    const int wn   = wave & 1;          // 0..1 : 64-col slice of tile

    const int blockM = blockIdx.y * TILE_M;
    const int blockN = blockIdx.x * TILE_N;

    // Staging: each thread owns three 16-float half-rows per K-slab
    //   A rows: ldRow and ldRow+128;  B row: ldRow
    const int ldRow = tid >> 1;         // 0..127
    const int ldCol = (tid & 1) << 4;   // 0 or 16

    const float* aPtr = A  + (size_t)(blockM + ldRow) * D_IN + ldCol;
    const float* bPtr = Bm + (size_t)(blockN + ldRow) * D_IN + ldCol;
    const int aOff0 = ldRow * LDSS + ldCol;
    const int aOff1 = (ldRow + 128) * LDSS + ldCol;
    const int bOff  = ldRow * LDSS + ldCol;

    // Per-lane fragment geometry (ISA 16-bit layouts, wave32)
    const int fm  = lane & 15;          // row (A) / col (B) within 16-tile
    const int aC0 = (lane >> 4) << 3;   // A: K chunks at {aC0, aC0+16}
    const int bK0 = (lane >> 4) << 3;   // B: K chunks at {bK0, bK0+16}

    v8f acc[4][4];
#pragma unroll
    for (int t = 0; t < 4; ++t)
#pragma unroll
        for (int u = 0; u < 4; ++u)
#pragma unroll
            for (int i = 0; i < 8; ++i)
                acc[t][u][i] = 0.0f;

    v4f aReg[2][4], bReg[4];

    // ---- prologue: load slab 0, stage to buffer 0, load slab 1 ----
#pragma unroll
    for (int i = 0; i < 4; ++i) {
        aReg[0][i] = *(const v4f*)(aPtr + 4 * i);
        aReg[1][i] = *(const v4f*)(aPtr + (size_t)128 * D_IN + 4 * i);
        bReg[i]    = *(const v4f*)(bPtr + 4 * i);
    }
    {
        _Float16* as0 = &As[0][aOff0];
        _Float16* as1 = &As[0][aOff1];
        _Float16* bs  = &Bs[0][bOff];
#pragma unroll
        for (int r = 0; r < 2; ++r) {
            v4h h0 = __builtin_convertvector(aReg[r][0], v4h);
            v4h h1 = __builtin_convertvector(aReg[r][1], v4h);
            v4h h2 = __builtin_convertvector(aReg[r][2], v4h);
            v4h h3 = __builtin_convertvector(aReg[r][3], v4h);
            _Float16* d = r ? as1 : as0;
            *(v8h*)(d)     = __builtin_shufflevector(h0, h1, 0,1,2,3,4,5,6,7);
            *(v8h*)(d + 8) = __builtin_shufflevector(h2, h3, 0,1,2,3,4,5,6,7);
        }
        v4h h0 = __builtin_convertvector(bReg[0], v4h);
        v4h h1 = __builtin_convertvector(bReg[1], v4h);
        v4h h2 = __builtin_convertvector(bReg[2], v4h);
        v4h h3 = __builtin_convertvector(bReg[3], v4h);
        *(v8h*)(bs)     = __builtin_shufflevector(h0, h1, 0,1,2,3,4,5,6,7);
        *(v8h*)(bs + 8) = __builtin_shufflevector(h2, h3, 0,1,2,3,4,5,6,7);
    }
#pragma unroll
    for (int i = 0; i < 4; ++i) {
        aReg[0][i] = *(const v4f*)(aPtr + TILE_K + 4 * i);
        aReg[1][i] = *(const v4f*)(aPtr + (size_t)128 * D_IN + TILE_K + 4 * i);
        bReg[i]    = *(const v4f*)(bPtr + TILE_K + 4 * i);
    }
    __syncthreads();

    // ---- main loop: one barrier per K-step ----
    for (int kt = 0; kt < KSTEPS; ++kt) {
        const int cur = kt & 1;

        // Stage slab kt+1 (already in regs) into the other buffer.
        if (kt + 1 < KSTEPS) {
            _Float16* as0 = &As[cur ^ 1][aOff0];
            _Float16* as1 = &As[cur ^ 1][aOff1];
            _Float16* bs  = &Bs[cur ^ 1][bOff];
#pragma unroll
            for (int r = 0; r < 2; ++r) {
                v4h h0 = __builtin_convertvector(aReg[r][0], v4h);
                v4h h1 = __builtin_convertvector(aReg[r][1], v4h);
                v4h h2 = __builtin_convertvector(aReg[r][2], v4h);
                v4h h3 = __builtin_convertvector(aReg[r][3], v4h);
                _Float16* d = r ? as1 : as0;
                *(v8h*)(d)     = __builtin_shufflevector(h0, h1, 0,1,2,3,4,5,6,7);
                *(v8h*)(d + 8) = __builtin_shufflevector(h2, h3, 0,1,2,3,4,5,6,7);
            }
            v4h h0 = __builtin_convertvector(bReg[0], v4h);
            v4h h1 = __builtin_convertvector(bReg[1], v4h);
            v4h h2 = __builtin_convertvector(bReg[2], v4h);
            v4h h3 = __builtin_convertvector(bReg[3], v4h);
            *(v8h*)(bs)     = __builtin_shufflevector(h0, h1, 0,1,2,3,4,5,6,7);
            *(v8h*)(bs + 8) = __builtin_shufflevector(h2, h3, 0,1,2,3,4,5,6,7);
        }

        // Prefetch slab kt+2 from global (overlaps with WMMA below).
        if (kt + 2 < KSTEPS) {
            const size_t ko = (size_t)(kt + 2) * TILE_K;
#pragma unroll
            for (int i = 0; i < 4; ++i) {
                aReg[0][i] = *(const v4f*)(aPtr + ko + 4 * i);
                aReg[1][i] = *(const v4f*)(aPtr + (size_t)128 * D_IN + ko + 4 * i);
                bReg[i]    = *(const v4f*)(bPtr + ko + 4 * i);
            }
        }

        // Compute 64x64 per wave from buffer `cur`: 16 WMMAs, 8 fragment loads.
        v16h af[4];
#pragma unroll
        for (int t = 0; t < 4; ++t) {
            const _Float16* p = &As[cur][(wm * 64 + t * 16 + fm) * LDSS];
            v8h lo = *(const v8h*)(p + aC0);
            v8h hi = *(const v8h*)(p + aC0 + 16);
            af[t] = __builtin_shufflevector(lo, hi, 0,1,2,3,4,5,6,7,8,9,10,11,12,13,14,15);
        }
#pragma unroll
        for (int u = 0; u < 4; ++u) {
            const _Float16* p = &Bs[cur][(wn * 64 + u * 16 + fm) * LDSS + bK0];
            v8h lo = *(const v8h*)(p);
            v8h hi = *(const v8h*)(p + 16);
            v16h bf = __builtin_shufflevector(lo, hi, 0,1,2,3,4,5,6,7,8,9,10,11,12,13,14,15);
#pragma unroll
            for (int t = 0; t < 4; ++t)
                acc[t][u] = __builtin_amdgcn_wmma_f32_16x16x32_f16(
                    false, af[t], false, bf, (short)0, acc[t][u], false, false);
        }

        __syncthreads();
    }

    // Epilogue: scale by boost[n], store f32.
    // C/D layout: VGPR i -> row (lane<16 ? i : 8+i), col = lane&15 within 16x16 tile.
    const int rowOff = (lane >> 4) << 3;  // 0 or 8
#pragma unroll
    for (int u = 0; u < 4; ++u) {
        const int n   = blockN + wn * 64 + u * 16 + fm;
        const float b = boost[n];
#pragma unroll
        for (int t = 0; t < 4; ++t) {
            const int mBase = blockM + wm * 64 + t * 16 + rowOff;
#pragma unroll
            for (int i = 0; i < 8; ++i)
                C[(size_t)(mBase + i) * N_COLS + n] = acc[t][u][i] * b;
        }
    }
}

// ---------------------------------------------------------------------------
// Kernel 2: per-row exact k-th-largest via 4-pass byte radix select on float
// bit patterns (all values >= 0 so uint order == float order), then mask with
// v >= threshold (tie-inclusive, matching jax top_k threshold semantics).
// ---------------------------------------------------------------------------
__global__ __launch_bounds__(256)
void htm_topk_mask(float* __restrict__ C, const int* __restrict__ numActive)
{
    __shared__ unsigned vals[N_COLS];   // 8 KB
    __shared__ unsigned hist[256];
    __shared__ unsigned sPrefix, sKK;

    const int tid = threadIdx.x;
    float* rowPtr = C + (size_t)blockIdx.x * N_COLS;

#pragma unroll
    for (int i = 0; i < N_COLS / 256; ++i)
        vals[tid + 256 * i] = __float_as_uint(rowPtr[tid + 256 * i]);

    if (tid == 0) {
        int k = numActive[0];
        if (k > N_COLS) k = N_COLS;
        if (k < 1) k = 1;
        sPrefix = 0u;
        sKK = (unsigned)k;
    }
    __syncthreads();

#pragma unroll
    for (int shift = 24; shift >= 0; shift -= 8) {
        hist[tid] = 0u;
        __syncthreads();
        const unsigned prefix = sPrefix;
        const unsigned mask = (shift == 24) ? 0u : (0xFFFFFFFFu << (shift + 8));
#pragma unroll
        for (int i = 0; i < N_COLS / 256; ++i) {
            unsigned v = vals[tid + 256 * i];
            if ((v & mask) == prefix)
                atomicAdd(&hist[(v >> shift) & 255u], 1u);
        }
        __syncthreads();
        if (tid == 0) {
            unsigned kk = sKK, cum = 0u;
            for (int b = 255; b >= 0; --b) {
                unsigned c = hist[b];
                if (cum + c >= kk) {
                    sKK = kk - cum;
                    sPrefix = prefix | ((unsigned)b << shift);
                    break;
                }
                cum += c;
            }
        }
        __syncthreads();
    }

    const unsigned thrBits = sPrefix;
#pragma unroll
    for (int i = 0; i < N_COLS / 256; ++i) {
        unsigned v = vals[tid + 256 * i];
        rowPtr[tid + 256 * i] = (v >= thrBits) ? __uint_as_float(v) : 0.0f;
    }
}

// ---------------------------------------------------------------------------
extern "C" void kernel_launch(void* const* d_in, const int* in_sizes, int n_in,
                              void* d_out, int out_size, void* d_ws, size_t ws_size,
                              hipStream_t stream)
{
    const float* input = (const float*)d_in[0];   // (2, 2048, 4096) f32
    const float* conn  = (const float*)d_in[1];   // (2048, 4096)    f32 (0/1)
    const float* boost = (const float*)d_in[2];   // (2048,)         f32
    const int*   nact  = (const int*)d_in[3];     // scalar int
    float* out = (float*)d_out;                   // (2, 2048, 2048) f32

    dim3 grid(N_COLS / TILE_N, M_ROWS / TILE_M);  // (16, 16)
    htm_overlap_gemm<<<grid, dim3(256), 0, stream>>>(input, conn, boost, out);
    htm_topk_mask<<<dim3(M_ROWS), dim3(256), 0, stream>>>(out, nact);

    (void)in_sizes; (void)n_in; (void)out_size; (void)d_ws; (void)ws_size;
}
